// BidirectionalCrossAttention_47227460387469
// MI455X (gfx1250) — compile-verified, matrix operand order checked by reading, not run on hardware
//
#include <hip/hip_runtime.h>
#include <hip/hip_bf16.h>

// ---------------------------------------------------------------------------
// Shapes
// ---------------------------------------------------------------------------
#define BATCH 4
#define CH    64
#define HW    4096            // 64*64
#define NTILE (HW / 16)       // 256 tiles of 16 along i or j

typedef __attribute__((ext_vector_type(16))) _Float16 v16h;
typedef __attribute__((ext_vector_type(8)))  _Float16 v8h;
typedef __attribute__((ext_vector_type(8)))  float    v8f;

#define WMMA_F32_F16(A, B, Cacc) \
  __builtin_amdgcn_wmma_f32_16x16x32_f16(false, (A), false, (B), (short)0, (Cacc), false, false)

// ---------------------------------------------------------------------------
// Fragment loaders.  Row-major (rows x ld) f16 buffer.
// 16-bit A 16x32 lane map (ISA 7.12.2): lane<16 -> row=lane, K in
// {c0..c0+7} u {c0+16..c0+23}; lane>=16 -> row=lane-16, K+8.  The B 32x16
// map is the same with "row" meaning the N column.  Two b128 loads per lane.
// ---------------------------------------------------------------------------
__device__ __forceinline__ const _Float16* frag_ptr(const _Float16* base, int ld,
                                                    int row0, int c0, int lane) {
  int r   = row0 + (lane & 15);
  int kph = (lane >> 4) << 3;
  return base + (size_t)r * (size_t)ld + (size_t)(c0 + kph);
}

__device__ __forceinline__ v16h load_frag(const _Float16* p) {
  v8h lo = *(const v8h*)(p);
  v8h hi = *(const v8h*)(p + 16);
  v16h r;
#pragma unroll
  for (int t = 0; t < 8; ++t) { r[t] = lo[t]; r[t + 8] = hi[t]; }
  return r;
}

// ---------------------------------------------------------------------------
// Kernel 1: six 1x1 projections -> f16 staging buffers.
//   qT[d][b][j][c] (q pre-scaled by 1/8), kT[d][b][i][c], v[d][b][c][i]
// proj index: 0 q1(x_s2,wq1) 1 k1(x_dem,wk1) 2 v1(x_dem,wv1)
//             3 q2(x_dem,wq2) 4 k2(x_s2,wk2) 5 v2(x_s2,wv2)
// ---------------------------------------------------------------------------
__global__ __launch_bounds__(256) void proj_kernel(
    const float* __restrict__ xs2, const float* __restrict__ xdem,
    const float* __restrict__ wq1, const float* __restrict__ bq1,
    const float* __restrict__ wk1, const float* __restrict__ bk1,
    const float* __restrict__ wv1, const float* __restrict__ bv1,
    const float* __restrict__ wq2, const float* __restrict__ bq2,
    const float* __restrict__ wk2, const float* __restrict__ bk2,
    const float* __restrict__ wv2, const float* __restrict__ bv2,
    _Float16* __restrict__ qT, _Float16* __restrict__ kT, _Float16* __restrict__ vv) {
  int p    = blockIdx.x * 256 + threadIdx.x;   // 0..4095
  int o    = blockIdx.y;                       // 0..63
  int z    = blockIdx.z;                       // proj*4 + b
  int proj = z >> 2;
  int b    = z & 3;

  const float* X[6]  = {xs2, xdem, xdem, xdem, xs2, xs2};
  const float* W[6]  = {wq1, wk1, wv1, wq2, wk2, wv2};
  const float* BS[6] = {bq1, bk1, bv1, bq2, bk2, bv2};

  const float* xb = X[proj] + ((size_t)b * CH) * HW + p;
  const float* w  = W[proj] + o * CH;
  float acc = BS[proj][o];
#pragma unroll 8
  for (int i = 0; i < CH; ++i) acc += w[i] * xb[(size_t)i * HW];

  int d = (proj < 3) ? 0 : 1;
  int kind = proj - 3 * d;                     // 0=q 1=k 2=v
  size_t img = (size_t)(d * BATCH + b);
  if (kind == 0)      qT[(img * HW + p) * CH + o] = (_Float16)(acc * 0.125f);
  else if (kind == 1) kT[(img * HW + p) * CH + o] = (_Float16)acc;
  else                vv[(img * CH + o) * HW + p] = (_Float16)acc;
}

// ---------------------------------------------------------------------------
// Kernel 2: softmax row statistics.  One wave per (d,b,i-tile of 16 rows).
// Streams all 256 j-tiles twice (max pass, sum pass); logits via WMMA.
// C/D fragment: slot t on lane L holds row M=t+8*(L>>4), col N=L&15.
// ---------------------------------------------------------------------------
__global__ __launch_bounds__(32) void attn_stats_kernel(
    const _Float16* __restrict__ qT, const _Float16* __restrict__ kT,
    float* __restrict__ mOut, float* __restrict__ lOut) {
  int it = blockIdx.x, b = blockIdx.y, d = blockIdx.z;
  int lane = threadIdx.x;
  size_t img = (size_t)(d * BATCH + b);
  const _Float16* qb = qT + img * (size_t)HW * CH;
  const _Float16* kb = kT + img * (size_t)HW * CH;
  int row0 = it * 16;

  v16h a0 = load_frag(frag_ptr(kb, CH, row0, 0,  lane));
  v16h a1 = load_frag(frag_ptr(kb, CH, row0, 32, lane));

  float m[8];
#pragma unroll
  for (int t = 0; t < 8; ++t) m[t] = -3.0e38f;

  for (int jt = 0; jt < NTILE; ++jt) {                 // pass 1: row max
    v16h b0 = load_frag(frag_ptr(qb, CH, jt * 16, 0,  lane));
    v16h b1 = load_frag(frag_ptr(qb, CH, jt * 16, 32, lane));
    v8f c = {};
    c = WMMA_F32_F16(a0, b0, c);
    c = WMMA_F32_F16(a1, b1, c);
#pragma unroll
    for (int t = 0; t < 8; ++t) m[t] = fmaxf(m[t], c[t]);
  }
  // reduce max across the 16 lanes of each half (xor stays inside the half)
#pragma unroll
  for (int t = 0; t < 8; ++t)
    for (int mask = 1; mask < 16; mask <<= 1)
      m[t] = fmaxf(m[t], __shfl_xor(m[t], mask, 32));

  float l[8];
#pragma unroll
  for (int t = 0; t < 8; ++t) l[t] = 0.0f;

  for (int jt = 0; jt < NTILE; ++jt) {                 // pass 2: sum of exp
    v16h b0 = load_frag(frag_ptr(qb, CH, jt * 16, 0,  lane));
    v16h b1 = load_frag(frag_ptr(qb, CH, jt * 16, 32, lane));
    v8f c = {};
    c = WMMA_F32_F16(a0, b0, c);
    c = WMMA_F32_F16(a1, b1, c);
#pragma unroll
    for (int t = 0; t < 8; ++t) l[t] += __expf(c[t] - m[t]);
  }
#pragma unroll
  for (int t = 0; t < 8; ++t)
    for (int mask = 1; mask < 16; mask <<= 1)
      l[t] += __shfl_xor(l[t], mask, 32);

  if ((lane & 15) == 0) {
    int hi = (lane >> 4) << 3;
    size_t base = img * HW + row0 + hi;
#pragma unroll
    for (int t = 0; t < 8; ++t) {
      mOut[base + t] = m[t];
      lOut[base + t] = 1.0f / l[t];
    }
  }
}

// ---------------------------------------------------------------------------
// Kernel 3: attention output.  One wave per (d,b,j-tile of 16 cols).
// Streams i in 32-row tiles; rebuilds logits with WMMA, exps in registers,
// reshapes p to an f16 B-fragment via LDS (one packed ds_store_b128 per
// half-tile), accumulates V x p with WMMA.
// ---------------------------------------------------------------------------
__global__ __launch_bounds__(32) void attn_out_kernel(
    const _Float16* __restrict__ qT, const _Float16* __restrict__ kT,
    const _Float16* __restrict__ vv, const float* __restrict__ mIn,
    const float* __restrict__ lIn, float* __restrict__ outAttn) {
  __shared__ __align__(16) _Float16 pT[16 * 32];       // [N=16][K=32]

  int jt = blockIdx.x, b = blockIdx.y, d = blockIdx.z;
  int lane = threadIdx.x;
  size_t img = (size_t)(d * BATCH + b);
  const _Float16* qb = qT + img * (size_t)HW * CH;
  const _Float16* kb = kT + img * (size_t)HW * CH;
  const _Float16* vb = vv + img * (size_t)CH * HW;
  const float* mrow = mIn + img * HW;
  const float* lrow = lIn + img * HW;
  int j0 = jt * 16;
  int n  = lane & 15;
  int hi = (lane >> 4) << 3;

  v16h q0 = load_frag(frag_ptr(qb, CH, j0, 0,  lane));
  v16h q1 = load_frag(frag_ptr(qb, CH, j0, 32, lane));

  v8f oacc[4];
#pragma unroll
  for (int mt = 0; mt < 4; ++mt) oacc[mt] = (v8f){};

  for (int it = 0; it < HW / 32; ++it) {
    int i0 = it * 32;
    if (it + 1 < HW / 32)
      __builtin_prefetch(frag_ptr(kb, CH, i0 + 32, 0, lane), 0, 3);
#pragma unroll
    for (int h = 0; h < 2; ++h) {
      int r0 = i0 + 16 * h;
      v16h k0 = load_frag(frag_ptr(kb, CH, r0, 0,  lane));
      v16h k1 = load_frag(frag_ptr(kb, CH, r0, 32, lane));
      v8f s = {};
      s = WMMA_F32_F16(k0, q0, s);
      s = WMMA_F32_F16(k1, q1, s);
      // pack this lane's 8 probabilities; their LDS slots are contiguous
      v8h pk;
#pragma unroll
      for (int t = 0; t < 8; ++t) {
        int i = r0 + t + hi;                           // global row index
        pk[t] = (_Float16)(__expf(s[t] - mrow[i]) * lrow[i]);
      }
      *(v8h*)&pT[n * 32 + 16 * h + hi] = pk;           // one ds_store_b128
    }
    __syncthreads();
    v16h pf = load_frag(&pT[n * 32 + hi]);             // B frag 32x16
#pragma unroll
    for (int mt = 0; mt < 4; ++mt) {
      v16h va = load_frag(frag_ptr(vb, HW, mt * 16, i0, lane));
      oacc[mt] = WMMA_F32_F16(va, pf, oacc[mt]);
    }
    __syncthreads();
  }

  // store out[c, j] (f32)
#pragma unroll
  for (int mt = 0; mt < 4; ++mt)
#pragma unroll
    for (int t = 0; t < 8; ++t) {
      int c = mt * 16 + t + hi;
      outAttn[(img * CH + c) * HW + (j0 + n)] = oacc[mt][t];
    }
}

// ---------------------------------------------------------------------------
// Kernel 4: residual add + LayerNorm over channels, in place on attn buffer.
// ---------------------------------------------------------------------------
__global__ __launch_bounds__(256) void ln_kernel(
    float* __restrict__ attn, const float* __restrict__ xs2,
    const float* __restrict__ xdem,
    const float* __restrict__ lnsw, const float* __restrict__ lnsb,
    const float* __restrict__ lndw, const float* __restrict__ lndb) {
  int p = blockIdx.x * 256 + threadIdx.x;
  int b = blockIdx.y, d = blockIdx.z;
  size_t img = (size_t)(d * BATCH + b);
  const float* xa = (d == 0) ? xs2 : xdem;
  const float* g  = (d == 0) ? lnsw : lndw;
  const float* bb = (d == 0) ? lnsb : lndb;
  float* buf = attn + img * (size_t)CH * HW;

  float vals[CH];
  float s = 0.0f, s2 = 0.0f;
#pragma unroll 8
  for (int c = 0; c < CH; ++c) {
    float v = buf[(size_t)c * HW + p] + xa[((size_t)b * CH + c) * HW + p];
    vals[c] = v; s += v; s2 += v * v;
  }
  float mu = s * (1.0f / CH);
  float var = s2 * (1.0f / CH) - mu * mu;
  float rs = rsqrtf(var + 1e-5f);
#pragma unroll 8
  for (int c = 0; c < CH; ++c)
    buf[(size_t)c * HW + p] = (vals[c] - mu) * rs * g[c] + bb[c];
}

// ---------------------------------------------------------------------------
// Kernel 5: 3x3 conv, 128 -> 64 channels, SAME padding.
// ---------------------------------------------------------------------------
__global__ __launch_bounds__(256) void conv3_kernel(
    const float* __restrict__ attn, const float* __restrict__ fw1,
    const float* __restrict__ fb1, float* __restrict__ y) {
  int p = blockIdx.x * 256 + threadIdx.x;
  int o = blockIdx.y, b = blockIdx.z;
  int h = p >> 6, w = p & 63;

  float acc = fb1[o];
  for (int ic = 0; ic < 2 * CH; ++ic) {
    int d = (ic < CH) ? 0 : 1;
    int c = ic - d * CH;
    const float* src = attn + ((size_t)(d * BATCH + b) * CH + c) * HW;
    const float* wp  = fw1 + ((size_t)o * (2 * CH) + ic) * 9;
#pragma unroll
    for (int kh = 0; kh < 3; ++kh) {
      int hh = h + kh - 1;
      if (hh < 0 || hh > 63) continue;
#pragma unroll
      for (int kw = 0; kw < 3; ++kw) {
        int ww = w + kw - 1;
        if (ww < 0 || ww > 63) continue;
        acc += wp[kh * 3 + kw] * src[hh * 64 + ww];
      }
    }
  }
  y[((size_t)b * CH + o) * HW + p] = acc;
}

// ---------------------------------------------------------------------------
// Kernel 6: BatchNorm statistics over (B,H,W), one block per channel.
// ---------------------------------------------------------------------------
__global__ __launch_bounds__(256) void bnstats_kernel(
    const float* __restrict__ y, float* __restrict__ bnMu, float* __restrict__ bnRs) {
  __shared__ float rs[256], rs2[256];
  int o = blockIdx.x, tid = threadIdx.x;
  float s = 0.0f, s2 = 0.0f;
  for (int idx = tid; idx < BATCH * HW; idx += 256) {
    int b = idx >> 12, p = idx & (HW - 1);
    float v = y[((size_t)b * CH + o) * HW + p];
    s += v; s2 += v * v;
  }
  rs[tid] = s; rs2[tid] = s2;
  __syncthreads();
  for (int st = 128; st > 0; st >>= 1) {
    if (tid < st) { rs[tid] += rs[tid + st]; rs2[tid] += rs2[tid + st]; }
    __syncthreads();
  }
  if (tid == 0) {
    float inv = 1.0f / (BATCH * HW);
    float mu = rs[0] * inv;
    float var = rs2[0] * inv - mu * mu;
    bnMu[o] = mu;
    bnRs[o] = rsqrtf(var + 1e-5f);
  }
}

// ---------------------------------------------------------------------------
// Kernel 7: fused BN-apply + ReLU + final 1x1 conv.
// ---------------------------------------------------------------------------
__global__ __launch_bounds__(256) void final_kernel(
    const float* __restrict__ y, const float* __restrict__ bnMu,
    const float* __restrict__ bnRs, const float* __restrict__ bng,
    const float* __restrict__ bnb, const float* __restrict__ fw2,
    const float* __restrict__ fb2, float* __restrict__ out) {
  int p = blockIdx.x * 256 + threadIdx.x;
  int o = blockIdx.y, b = blockIdx.z;
  float acc = fb2[o];
#pragma unroll 4
  for (int i = 0; i < CH; ++i) {
    float v = y[((size_t)b * CH + i) * HW + p];
    v = (v - bnMu[i]) * bnRs[i] * bng[i] + bnb[i];
    v = fmaxf(v, 0.0f);
    acc += fw2[o * CH + i] * v;
  }
  out[((size_t)b * CH + o) * HW + p] = acc;
}

// ---------------------------------------------------------------------------
// Launch
// ---------------------------------------------------------------------------
extern "C" void kernel_launch(void* const* d_in, const int* in_sizes, int n_in,
                              void* d_out, int out_size, void* d_ws, size_t ws_size,
                              hipStream_t stream) {
  const float* xs2  = (const float*)d_in[0];
  const float* xdem = (const float*)d_in[1];
  const float* wq1 = (const float*)d_in[2];  const float* bq1 = (const float*)d_in[3];
  const float* wk1 = (const float*)d_in[4];  const float* bk1 = (const float*)d_in[5];
  const float* wv1 = (const float*)d_in[6];  const float* bv1 = (const float*)d_in[7];
  const float* wq2 = (const float*)d_in[8];  const float* bq2 = (const float*)d_in[9];
  const float* wk2 = (const float*)d_in[10]; const float* bk2 = (const float*)d_in[11];
  const float* wv2 = (const float*)d_in[12]; const float* bv2 = (const float*)d_in[13];
  const float* lnsw = (const float*)d_in[14]; const float* lnsb = (const float*)d_in[15];
  const float* lndw = (const float*)d_in[16]; const float* lndb = (const float*)d_in[17];
  const float* fw1 = (const float*)d_in[18]; const float* fb1 = (const float*)d_in[19];
  const float* bng = (const float*)d_in[20]; const float* bnb = (const float*)d_in[21];
  const float* fw2 = (const float*)d_in[22]; const float* fb2 = (const float*)d_in[23];
  float* out = (float*)d_out;

  char* ws = (char*)d_ws;
  size_t off = 0;
  auto take = [&](size_t bytes) {
    char* p = ws + off;
    off = (off + bytes + 255) & ~(size_t)255;
    return p;
  };
  const size_t nimg = 2 * BATCH;
  _Float16* qT  = (_Float16*)take(nimg * HW * CH * sizeof(_Float16));   // 4 MB
  _Float16* kT  = (_Float16*)take(nimg * HW * CH * sizeof(_Float16));   // 4 MB
  _Float16* vv  = (_Float16*)take(nimg * CH * HW * sizeof(_Float16));   // 4 MB
  float* mBuf   = (float*)take(nimg * HW * sizeof(float));              // 128 KB
  float* lBuf   = (float*)take(nimg * HW * sizeof(float));              // 128 KB
  float* attn   = (float*)take(nimg * CH * HW * sizeof(float));         // 8 MB
  float* ybuf   = (float*)take((size_t)BATCH * CH * HW * sizeof(float));// 4 MB
  float* bnMu   = (float*)take(CH * sizeof(float));
  float* bnRs   = (float*)take(CH * sizeof(float));

  proj_kernel<<<dim3(HW / 256, CH, 24), 256, 0, stream>>>(
      xs2, xdem, wq1, bq1, wk1, bk1, wv1, bv1, wq2, bq2, wk2, bk2, wv2, bv2,
      qT, kT, vv);
  attn_stats_kernel<<<dim3(NTILE, BATCH, 2), 32, 0, stream>>>(qT, kT, mBuf, lBuf);
  attn_out_kernel<<<dim3(NTILE, BATCH, 2), 32, 0, stream>>>(qT, kT, vv, mBuf, lBuf, attn);
  ln_kernel<<<dim3(HW / 256, BATCH, 2), 256, 0, stream>>>(
      attn, xs2, xdem, lnsw, lnsb, lndw, lndb);
  conv3_kernel<<<dim3(HW / 256, CH, BATCH), 256, 0, stream>>>(attn, fw1, fb1, ybuf);
  bnstats_kernel<<<dim3(CH), 256, 0, stream>>>(ybuf, bnMu, bnRs);
  final_kernel<<<dim3(HW / 256, CH, BATCH), 256, 0, stream>>>(
      ybuf, bnMu, bnRs, bng, bnb, fw2, fb2, out);
}
